// LandmarkGate_42649025249599
// MI455X (gfx1250) — matrix operand order repaired and checked by reading, not compiled, not run on hardware
//
#include <hip/hip_runtime.h>
#include <hip/hip_bf16.h>
#include <hip/hip_fp16.h>

typedef __attribute__((ext_vector_type(16))) _Float16 v16h;
typedef __attribute__((ext_vector_type(8)))  _Float16 v8h;
typedef __attribute__((ext_vector_type(8)))  float    v8f;

#define BLOCK_SIZE 256
#define WAVES_PER_BLOCK (BLOCK_SIZE / 32)

// Cross-row (xor 16) combine on the VALU (v_permlanex16_b32, identity selects).
static __device__ __forceinline__ float xor16_sum(float p) {
#if __has_builtin(__builtin_amdgcn_permlanex16)
    int pi = __float_as_int(p);
    int qi = __builtin_amdgcn_permlanex16(pi, pi, 0x76543210, 0xFEDCBA98, false, false);
    return p + __int_as_float(qi);
#else
    return p + __shfl_xor(p, 16);
#endif
}

// Build one B column (feat^T for one edge) in this kernel's contiguous k order (k = 16*half + s, same as A):
// lanes 0..15 hold K=0..15 (src emb, dst emb), lanes 16..31 hold K=16..31
// (lm[src], lm[dst], zeros).
template <bool F16TAB>
static __device__ __forceinline__ v16h build_b(
    int hi, int si, int di,
    const float* __restrict__ nt_emb, const float* __restrict__ lm_mask,
    const _Float16* __restrict__ embh, const _Float16* __restrict__ lmh)
{
    v16h b;
#pragma unroll
    for (int j = 0; j < 16; ++j) b[j] = (_Float16)0.f;
    if (F16TAB) {
        if (hi == 0) {
            const v8h* t = (const v8h*)embh;
            v8h s = t[si];
            v8h d = t[di];
#pragma unroll
            for (int j = 0; j < 8; ++j) { b[j] = s[j]; b[8 + j] = d[j]; }
        } else {
            b[0] = lmh[si];
            b[1] = lmh[di];
        }
    } else {
        if (hi == 0) {
            const float4* t = (const float4*)nt_emb;
            float4 s0 = t[(long long)si * 2 + 0];
            float4 s1 = t[(long long)si * 2 + 1];
            float4 d0 = t[(long long)di * 2 + 0];
            float4 d1 = t[(long long)di * 2 + 1];
            b[0]  = (_Float16)s0.x; b[1]  = (_Float16)s0.y;
            b[2]  = (_Float16)s0.z; b[3]  = (_Float16)s0.w;
            b[4]  = (_Float16)s1.x; b[5]  = (_Float16)s1.y;
            b[6]  = (_Float16)s1.z; b[7]  = (_Float16)s1.w;
            b[8]  = (_Float16)d0.x; b[9]  = (_Float16)d0.y;
            b[10] = (_Float16)d0.z; b[11] = (_Float16)d0.w;
            b[12] = (_Float16)d1.x; b[13] = (_Float16)d1.y;
            b[14] = (_Float16)d1.z; b[15] = (_Float16)d1.w;
        } else {
            b[0] = (_Float16)lm_mask[si];
            b[1] = (_Float16)lm_mask[di];
        }
    }
    return b;
}

// Prep: convert embedding table and mask to f16 once per launch (L2-resident).
__global__ void LandmarkGate_cvt_kernel(const float* __restrict__ nt_emb,
                                        const float* __restrict__ lm_mask,
                                        _Float16* __restrict__ embh,
                                        _Float16* __restrict__ lmh,
                                        int nEmb, int nMask)
{
    int i      = blockIdx.x * blockDim.x + threadIdx.x;
    int stride = gridDim.x * blockDim.x;
    for (int j = i; j < nEmb; j += stride)  embh[j] = (_Float16)nt_emb[j];
    for (int j = i; j < nMask; j += stride) lmh[j]  = (_Float16)lm_mask[j];
}

// One wave = 32 edges (two 16-edge tiles) per iteration.
//   A (16x32 f16) = W1^T tiles (hidden rows 0..15 / 16..31), loop-invariant
//   B (32x16 f16) = feat^T columns (one per edge)
//   C (16x16 f32) = b1 bias, preloaded -> WMMA D arrives pre-biased
//   D (16x16 f32) = h^T: hidden units along accumulator VGPRs -> in-lane dot,
//   one v_permlanex16 per tile, one exp + one coalesced NT store per 32 edges.
template <bool F16TAB>
__global__ __launch_bounds__(BLOCK_SIZE)
void LandmarkGate_42649025249599_kernel(
    const float*     __restrict__ nt_emb,   // [N,8]
    const long long* __restrict__ e_src,    // [E]
    const long long* __restrict__ e_dst,    // [E]
    const float*     __restrict__ lm_mask,  // [N]
    const float*     __restrict__ W1,       // [18,32] row-major
    const float*     __restrict__ b1,       // [32]
    const float*     __restrict__ W2,       // [32]
    const float*     __restrict__ b2v,      // [1]
    const _Float16*  __restrict__ embh,     // [N,8] f16 (if F16TAB)
    const _Float16*  __restrict__ lmh,      // [N]   f16 (if F16TAB)
    float*           __restrict__ out,      // [E]
    long long E, int numPairs)
{
    const int lane = threadIdx.x & 31;
    const int col  = lane & 15;
    const int hi   = lane >> 4;

    // ---- A fragments: W1^T, hidden rows 0..15 (aLo) and 16..31 (aHi) ----
    v16h aLo, aHi;
#pragma unroll
    for (int j = 0; j < 16; ++j) {
        int k = hi * 16 + j;
        float wlo = 0.f, whi = 0.f;
        if (k < 18) {
            wlo = W1[k * 32 + col];
            whi = W1[k * 32 + 16 + col];
        }
        aLo[j] = (_Float16)wlo;
        aHi[j] = (_Float16)whi;
    }

    // ---- C fragments: layer-1 bias in C/D row layout (rows hi*8..hi*8+7) ----
    v8f cLo, cHi;
    float w2A[8], w2B[8];
#pragma unroll
    for (int r = 0; r < 8; ++r) {
        cLo[r] = b1[hi * 8 + r];
        cHi[r] = b1[16 + hi * 8 + r];
        w2A[r] = W2[hi * 8 + r];
        w2B[r] = W2[16 + hi * 8 + r];
    }
    const float bout = b2v[0];

    const int wavesTotal = gridDim.x * WAVES_PER_BLOCK;
    const int wave       = blockIdx.x * WAVES_PER_BLOCK + (threadIdx.x >> 5);

    for (int pair = wave; pair < numPairs; pair += wavesTotal) {
        const long long base = (long long)pair * 32;
        const long long e0   = base + col;
        const long long e1   = base + 16 + col;
        const long long ee0  = (e0 < E) ? e0 : 0;
        const long long ee1  = (e1 < E) ? e1 : 0;

        // Read-once int64 index streams (NT so they don't evict the tables).
        long long s0 = __builtin_nontemporal_load(e_src + ee0);
        long long d0 = __builtin_nontemporal_load(e_dst + ee0);
        long long s1 = __builtin_nontemporal_load(e_src + ee1);
        long long d1 = __builtin_nontemporal_load(e_dst + ee1);

        // Prefetch next pair's 256B of each index array.
        long long npair = (long long)pair + wavesTotal;
        if (npair < (long long)numPairs) {
            long long pe = npair * 32 + col * 2;
            __builtin_prefetch((hi == 0) ? (const void*)(e_src + pe)
                                         : (const void*)(e_dst + pe), 0, 0);
        }

        v16h bb0 = build_b<F16TAB>(hi, (int)s0, (int)d0, nt_emb, lm_mask, embh, lmh);
        v16h bb1 = build_b<F16TAB>(hi, (int)s1, (int)d1, nt_emb, lm_mask, embh, lmh);

        // ---- Layer 1: four independent WMMAs, bias folded into C ----
        v8f dLo0 = __builtin_amdgcn_wmma_f32_16x16x32_f16(false, aLo, false, bb0,
                                                          (short)0, cLo, false, false);
        v8f dHi0 = __builtin_amdgcn_wmma_f32_16x16x32_f16(false, aHi, false, bb0,
                                                          (short)0, cHi, false, false);
        v8f dLo1 = __builtin_amdgcn_wmma_f32_16x16x32_f16(false, aLo, false, bb1,
                                                          (short)0, cLo, false, false);
        v8f dHi1 = __builtin_amdgcn_wmma_f32_16x16x32_f16(false, aHi, false, bb1,
                                                          (short)0, cHi, false, false);

        // ---- Layer 2: relu + in-lane dot over this lane's 16 hidden rows ----
        float p0 = 0.f, p1 = 0.f;
#pragma unroll
        for (int r = 0; r < 8; ++r) {
            p0 += fmaxf(dLo0[r], 0.f) * w2A[r];
            p0 += fmaxf(dHi0[r], 0.f) * w2B[r];
            p1 += fmaxf(dLo1[r], 0.f) * w2A[r];
            p1 += fmaxf(dHi1[r], 0.f) * w2B[r];
        }

        // Full sums land in BOTH half-waves; low half stores tile0, high half
        // stores tile1 -> one exp + one coalesced 128B NT store per 32 edges.
        float tot0 = xor16_sum(p0);
        float tot1 = xor16_sum(p1);

        long long oe = base + (long long)(hi * 16 + col);
        if (oe < E) {
            float x = (hi ? tot1 : tot0) + bout;
            float y = 1.f / (1.f + __expf(-x));
            __builtin_nontemporal_store(y, out + oe);
        }
    }
}

extern "C" void kernel_launch(void* const* d_in, const int* in_sizes, int n_in,
                              void* d_out, int out_size, void* d_ws, size_t ws_size,
                              hipStream_t stream) {
    const float*     nt_emb     = (const float*)d_in[0];
    const long long* edge_index = (const long long*)d_in[1];   // int64 [2,E]
    const float*     lm_mask    = (const float*)d_in[2];
    const float*     W1         = (const float*)d_in[3];
    const float*     b1         = (const float*)d_in[4];
    const float*     W2         = (const float*)d_in[5];
    const float*     b2         = (const float*)d_in[6];
    float*           out        = (float*)d_out;

    const long long E = (long long)in_sizes[1] / 2;
    const long long* e_src = edge_index;
    const long long* e_dst = edge_index + E;

    const int nEmb  = in_sizes[0];      // N*8
    const int nMask = in_sizes[2];      // N

    int numPairs = (int)((E + 31) / 32);
    int blocksNeeded = (numPairs + WAVES_PER_BLOCK - 1) / WAVES_PER_BLOCK;
    int blocks = blocksNeeded < 4096 ? blocksNeeded : 4096;
    if (blocks < 1) blocks = 1;

    const size_t needWs = (size_t)nEmb * sizeof(_Float16)
                        + (size_t)nMask * sizeof(_Float16);

    if (ws_size >= needWs) {
        _Float16* embh = (_Float16*)d_ws;
        _Float16* lmh  = embh + nEmb;
        int cvtBlocks = (nEmb + BLOCK_SIZE - 1) / BLOCK_SIZE;
        if (cvtBlocks > 1024) cvtBlocks = 1024;
        LandmarkGate_cvt_kernel<<<cvtBlocks, BLOCK_SIZE, 0, stream>>>(
            nt_emb, lm_mask, embh, lmh, nEmb, nMask);
        LandmarkGate_42649025249599_kernel<true><<<blocks, BLOCK_SIZE, 0, stream>>>(
            nt_emb, e_src, e_dst, lm_mask, W1, b1, W2, b2, embh, lmh,
            out, E, numPairs);
    } else {
        LandmarkGate_42649025249599_kernel<false><<<blocks, BLOCK_SIZE, 0, stream>>>(
            nt_emb, e_src, e_dst, lm_mask, W1, b1, W2, b2,
            (const _Float16*)nullptr, (const _Float16*)nullptr,
            out, E, numPairs);
    }
}
